// TopKSparseAutoEncoder_20847771255393
// MI455X (gfx1250) — matrix operand order repaired and checked by reading, not compiled, not run on hardware
//
#include <hip/hip_runtime.h>
#include <hip/hip_bf16.h>

// ---------------------------------------------------------------------------
// Types for WMMA
// ---------------------------------------------------------------------------
typedef unsigned int u32;
typedef __attribute__((ext_vector_type(4))) u32 u32x4;
typedef __bf16 bf16_t;
typedef __attribute__((ext_vector_type(2)))  bf16_t v2bf;
typedef __attribute__((ext_vector_type(16))) bf16_t v16bf;
typedef __attribute__((ext_vector_type(8)))  float  v8f;

union Frag {
    u32x4 q[2];
    v16bf v;
};

// Hardware f32->bf16 conversion (v_cvt_pk_bf16_f32 class), not bit-twiddling.
__device__ __forceinline__ u32 pack_bf16(float lo, float hi) {
    union { v2bf v; u32 u; } r;
    r.v[0] = (bf16_t)lo;
    r.v[1] = (bf16_t)hi;
    return r.u;
}

// ---------------------------------------------------------------------------
// Kernel 1: decoder row 2-norms. norms[f] = ||W_dec[f,:]||_2. Memory-bound
// (512 MB stream ~= 22us at 23.3 TB/s).
// ---------------------------------------------------------------------------
__global__ __launch_bounds__(256) void sae_norms_kernel(
    const float* __restrict__ Wd, float* __restrict__ norms, int D) {
    __shared__ float red[256];
    const long f = blockIdx.x;
    const float* row = Wd + f * (long)D;
    float s = 0.f;
    for (int i = threadIdx.x; i < D; i += 256) {
        float x = row[i];
        s += x * x;
    }
    red[threadIdx.x] = s;
    __syncthreads();
    for (int off = 128; off > 0; off >>= 1) {
        if (threadIdx.x < off) red[threadIdx.x] += red[threadIdx.x + off];
        __syncthreads();
    }
    if (threadIdx.x == 0) norms[f] = sqrtf(red[0]);
}

// ---------------------------------------------------------------------------
// Kernel 2: encode GEMM + ReLU via V_WMMA_F32_16X16X32_BF16.
// act[M,N] = relu(A[M,K] @ B[K,N]).
// Block tile 128x128, K-tile 32, 256 threads = 8 waves.
// Wave grid 4x2; each wave owns a 32x64 tile = 2x4 WMMA fragments (8 WMMA/iter).
// Software pipeline: next K-tile's globals are fetched into registers before
// the WMMA block so HBM latency overlaps matrix math; fp32->bf16 conversion
// uses the hardware cvt path during the LDS store phase.
// ---------------------------------------------------------------------------
#define BM 128
#define BN 128
#define BK 32

__global__ __launch_bounds__(256) void sae_encode_wmma_kernel(
    const float* __restrict__ A, const float* __restrict__ Bm,
    float* __restrict__ act, int N, int K) {
    // packed bf16 pairs: sA[row][kdword 0..15], sB[col][kdword 0..15]
    __shared__ u32 sA[BM * 16];
    __shared__ u32 sB[BN * 16];

    const int tid  = threadIdx.x;
    const int lane = tid & 31;
    const int wave = tid >> 5;
    const int wr = wave >> 1;          // wave row 0..3  (32 rows each)
    const int wc = wave & 1;           // wave col 0..1  (64 cols each)
    const int rowBase = blockIdx.y * BM;
    const int colBase = blockIdx.x * BN;

    // A loader: thread -> (row aR, 16-col half aH); 16 consecutive floats.
    const int aR = tid & 127;
    const int aH = tid >> 7;
    // B loader: thread -> (K-pair bK2, 8-col group bNb); stores transposed.
    const int bK2 = tid & 15;
    const int bNb = tid >> 4;

    const float* aptr_base = A + (long)(rowBase + aR) * K + aH * 16;
    const float* bptr_base = Bm + (long)(2 * bK2) * N + colBase + bNb * 8;

    v8f c[2][4];
#pragma unroll
    for (int mi = 0; mi < 2; ++mi)
#pragma unroll
        for (int ni = 0; ni < 4; ++ni)
#pragma unroll
            for (int e = 0; e < 8; ++e) c[mi][ni][e] = 0.f;

    float a_reg[16];
    float b_reg0[8], b_reg1[8];

    // ---- prologue: fetch + stage K-tile 0
    {
        const float* ap = aptr_base;
        const float* b0 = bptr_base;
        const float* b1 = b0 + N;
#pragma unroll
        for (int j = 0; j < 16; ++j) a_reg[j] = ap[j];
#pragma unroll
        for (int n = 0; n < 8; ++n) { b_reg0[n] = b0[n]; b_reg1[n] = b1[n]; }
#pragma unroll
        for (int j = 0; j < 8; ++j)
            sA[aR * 16 + aH * 8 + j] = pack_bf16(a_reg[2 * j], a_reg[2 * j + 1]);
#pragma unroll
        for (int n = 0; n < 8; ++n)
            sB[(bNb * 8 + n) * 16 + bK2] = pack_bf16(b_reg0[n], b_reg1[n]);
    }
    __syncthreads();

    for (int k0 = 0; k0 < K; k0 += BK) {
        const bool more = (k0 + BK) < K;

        // ---- fetch next K-tile into registers (overlaps with WMMA below)
        if (more) {
            const float* ap = aptr_base + (k0 + BK);
            const float* b0 = bptr_base + (long)(k0 + BK) * N;
            const float* b1 = b0 + N;
#pragma unroll
            for (int j = 0; j < 16; ++j) a_reg[j] = ap[j];
#pragma unroll
            for (int n = 0; n < 8; ++n) { b_reg0[n] = b0[n]; b_reg1[n] = b1[n]; }
        }

        // ---- fragments from LDS
        const int fr  = lane & 15;
        const int akd = (lane >> 4) * 8;   // A: 16 consecutive K per lane half
        const int bkd = (lane >> 4) * 8;   // B: 16 consecutive K per lane half

        Frag af[2], bf[4];
#pragma unroll
        for (int mi = 0; mi < 2; ++mi) {
            const u32* base = &sA[(wr * 32 + mi * 16 + fr) * 16 + akd];
            af[mi].q[0] = *(const u32x4*)(base);
            af[mi].q[1] = *(const u32x4*)(base + 4);
        }
#pragma unroll
        for (int ni = 0; ni < 4; ++ni) {
            const u32* base = &sB[(wc * 64 + ni * 16 + fr) * 16 + bkd];
            bf[ni].q[0] = *(const u32x4*)(base);
            bf[ni].q[1] = *(const u32x4*)(base + 4);
        }

#pragma unroll
        for (int mi = 0; mi < 2; ++mi)
#pragma unroll
            for (int ni = 0; ni < 4; ++ni)
                c[mi][ni] = __builtin_amdgcn_wmma_f32_16x16x32_bf16(
                    false, af[mi].v, false, bf[ni].v,
                    (short)0, c[mi][ni], false, false);

        __syncthreads();

        // ---- convert + stage the prefetched tile
        if (more) {
#pragma unroll
            for (int j = 0; j < 8; ++j)
                sA[aR * 16 + aH * 8 + j] = pack_bf16(a_reg[2 * j], a_reg[2 * j + 1]);
#pragma unroll
            for (int n = 0; n < 8; ++n)
                sB[(bNb * 8 + n) * 16 + bK2] = pack_bf16(b_reg0[n], b_reg1[n]);
            __syncthreads();
        }
    }

    // ---- epilogue: ReLU + store. C layout: VGPR v -> M = v + 8*(lane>=16), N = lane&15
    const int fr   = lane & 15;
    const int rsel = (lane >> 4) * 8;
#pragma unroll
    for (int mi = 0; mi < 2; ++mi)
#pragma unroll
        for (int ni = 0; ni < 4; ++ni) {
            const int col = colBase + wc * 64 + ni * 16 + fr;
#pragma unroll
            for (int v = 0; v < 8; ++v) {
                const int row = rowBase + wr * 32 + mi * 16 + rsel + v;
                float x = c[mi][ni][v];
                x = x > 0.f ? x : 0.f;
                act[(long)row * N + col] = x;
            }
        }
}

// ---------------------------------------------------------------------------
// Kernel 3: per-token top-k. One 1024-thread workgroup per token.
// The full gating row (F=32768 floats = 128 KB) lives in gfx1250's 320 KB LDS;
// 50 LDS-resident argmax passes. Also zero-fills and scatters the sparse row
// and emits topk indices (as float).
// ---------------------------------------------------------------------------
__global__ __launch_bounds__(1024) void sae_topk_kernel(
    const float* __restrict__ act, const float* __restrict__ norms,
    float* __restrict__ sparse, float* __restrict__ idx_out, int F, int k) {
    extern __shared__ float smem[];
    float* g  = smem;             // F gating values
    float* rv = smem + F;         // 1024 reduction values
    int*   ri = (int*)(rv + 1024);
    __shared__ int selIdx[64];

    const int  tid = threadIdx.x;
    const long b   = blockIdx.x;
    const float* arow = act + b * (long)F;
    float*       srow = sparse + b * (long)F;

    for (int i = tid; i < F; i += 1024) {
        g[i] = arow[i] * norms[i];
        srow[i] = 0.f;
    }
    __syncthreads();

    for (int it = 0; it < k; ++it) {
        float bv = -3.4e38f;
        int   bi = 0;
        for (int i = tid; i < F; i += 1024) {
            float v = g[i];
            if (v > bv) { bv = v; bi = i; }
        }
        rv[tid] = bv;
        ri[tid] = bi;
        __syncthreads();
        for (int off = 512; off > 0; off >>= 1) {
            if (tid < off && rv[tid + off] > rv[tid]) {
                rv[tid] = rv[tid + off];
                ri[tid] = ri[tid + off];
            }
            __syncthreads();
        }
        if (tid == 0) {
            int ix = ri[0];
            selIdx[it] = ix;
            g[ix] = -3.4e38f;
        }
        __syncthreads();
    }

    if (tid < k) {
        const int ix = selIdx[tid];
        idx_out[b * k + tid] = (float)ix;
        srow[ix] = arow[ix];   // activation value (not gating)
    }
}

// ---------------------------------------------------------------------------
// Kernel 4: sparse decode. recon[b,:] = b_dec + sum_i val_i * W_dec[idx_i,:].
// One 256-thread workgroup per token; only the k=50 selected rows are touched.
// ---------------------------------------------------------------------------
__global__ __launch_bounds__(256) void sae_decode_kernel(
    const float* __restrict__ sparse, const float* __restrict__ idx_out,
    const float* __restrict__ Wd, const float* __restrict__ bdec,
    float* __restrict__ recon, int D, int F, int k) {
    __shared__ int   sIdx[64];
    __shared__ float sVal[64];
    const int  tid = threadIdx.x;
    const long b   = blockIdx.x;

    if (tid < k) {
        const int ix = (int)idx_out[b * k + tid];
        sIdx[tid] = ix;
        sVal[tid] = sparse[b * (long)F + ix];
    }
    __syncthreads();

    float acc[16];
#pragma unroll
    for (int j = 0; j < 16; ++j) {
        const int d = tid + j * 256;
        acc[j] = (d < D) ? bdec[d] : 0.f;
    }
    for (int i = 0; i < k; ++i) {
        const float* wrow = Wd + (long)sIdx[i] * D;
        const float  v    = sVal[i];
#pragma unroll
        for (int j = 0; j < 16; ++j) {
            const int d = tid + j * 256;
            if (d < D) acc[j] += v * wrow[d];
        }
    }
#pragma unroll
    for (int j = 0; j < 16; ++j) {
        const int d = tid + j * 256;
        if (d < D) recon[b * (long)D + d] = acc[j];
    }
}

// ---------------------------------------------------------------------------
// Launch
// ---------------------------------------------------------------------------
extern "C" void kernel_launch(void* const* d_in, const int* in_sizes, int n_in,
                              void* d_out, int out_size, void* d_ws, size_t ws_size,
                              hipStream_t stream) {
    const float* hidden = (const float*)d_in[0];
    const float* W_enc  = (const float*)d_in[1];
    const float* W_dec  = (const float*)d_in[2];
    const float* b_dec  = (const float*)d_in[3];
    // d_in[4] is the device-side scalar k; k is recovered on the host from
    // out_size so launches stay graph-capture-safe.

    const int D = in_sizes[3];
    const int B = in_sizes[0] / D;
    const int F = in_sizes[1] / D;
    const long bk = (long)out_size - (long)B * D - 2L * (long)B * F;
    const int k = (int)(bk / B);   // 50 for the reference config

    float* out     = (float*)d_out;
    float* recon   = out;                           // [B, D]
    float* sparse  = recon + (long)B * D;           // [B, F]
    float* idx_out = sparse + (long)B * F;          // [B, k] as float
    float* act     = idx_out + (long)B * k;         // [B, F]
    float* norms   = (float*)d_ws;                  // [F]

    // 1) decoder norms
    sae_norms_kernel<<<F, 256, 0, stream>>>(W_dec, norms, D);

    // 2) encode GEMM + ReLU (bf16 WMMA)
    dim3 grid(F / BN, B / BM);
    sae_encode_wmma_kernel<<<grid, 256, 0, stream>>>(hidden, W_enc, act, F, D);

    // 3) top-k (gating row resident in LDS) + sparse scatter
    const size_t shmem = (size_t)F * sizeof(float) + 1024 * (sizeof(float) + sizeof(int));
    sae_topk_kernel<<<B, 1024, shmem, stream>>>(act, norms, sparse, idx_out, F, k);

    // 4) sparse decode
    sae_decode_kernel<<<B, 256, 0, stream>>>(sparse, idx_out, W_dec, b_dec, recon, D, F, k);
}
